// RNN_42082089566824
// MI455X (gfx1250) — matrix operand order repaired and checked by reading, not compiled, not run on hardware
//
#include <hip/hip_runtime.h>
#include <hip/hip_bf16.h>

typedef __attribute__((ext_vector_type(16))) __bf16 v16bf;
typedef __attribute__((ext_vector_type(8)))  float  v8f;

#define B_SZ   8192
#define D_INP  128
#define D_COND 3
#define H_SZ   1024
#define L_SZ   4
#define NQ     8
#define K_SZ   1024
#define DEC_IN (H_SZ + D_INP)      // 1152
#define INV_TAU (1.0f / 0.6f)

// ---------------------------------------------------------------- fragments
// A fragment (16x32 bf16, MxK): lane half h, m = lane%16,
//   elements e0..7 -> K = k0+8h .. +7 ; e8..15 -> K = k0+16+8h ..
__device__ __forceinline__ v16bf loadA16(const __bf16* base, int ld, int m0, int k0) {
  const int lane = threadIdx.x & 31;
  const int hf   = lane >> 4;
  const int m    = m0 + (lane & 15);
  const __bf16* p = base + (size_t)m * ld + k0 + 8 * hf;
  union { v16bf v; uint4 q[2]; } u;
  u.q[0] = *(const uint4*)(p);
  u.q[1] = *(const uint4*)(p + 16);
  return u.v;
}

// B fragment (32x16 bf16, KxN): lane half h, n = lane%16,
//   elements e0..7 -> K = k0+8h .. +7 ; e8..15 -> K = k0+16+8h ..
__device__ __forceinline__ v16bf loadB16(const __bf16* base, int ld, int n0, int k0) {
  const int lane = threadIdx.x & 31;
  const int hf   = lane >> 4;
  const int n    = n0 + (lane & 15);
  const __bf16* p = base + (size_t)n * ld + k0 + 8 * hf;
  union { v16bf v; uint4 q[2]; } u;
  u.q[0] = *(const uint4*)(p);
  u.q[1] = *(const uint4*)(p + 16);
  return u.v;
}

__device__ __forceinline__ v8f wmma_bf16(v16bf a, v16bf b, v8f c) {
  return __builtin_amdgcn_wmma_f32_16x16x32_bf16(false, a, false, b, (short)0, c,
                                                 false, false);
}

__device__ __forceinline__ float sigmoidf_(float x) {
  return 1.0f / (1.0f + __expf(-x));
}

// ---------------------------------------------------------------- converters
__global__ void cvt_bf16_kernel(const float* __restrict__ src,
                                __bf16* __restrict__ dst, size_t n) {
  size_t i = (size_t)blockIdx.x * blockDim.x + threadIdx.x;
  size_t stride = (size_t)gridDim.x * blockDim.x;
  for (; i < n; i += stride) dst[i] = (__bf16)src[i];
}

// E_eff (NQ,K,128) f32 -> ET (NQ,128,K) bf16 (reduction dim contiguous)
__global__ void transE_kernel(const float* __restrict__ E, __bf16* __restrict__ ET) {
  size_t i = (size_t)blockIdx.x * blockDim.x + threadIdx.x;
  if (i >= (size_t)NQ * K_SZ * D_INP) return;
  int d = (int)(i % D_INP);
  size_t t = i / D_INP;
  int k = (int)(t % K_SZ);
  int q = (int)(t / K_SZ);
  ET[((size_t)q * D_INP + d) * K_SZ + k] = (__bf16)E[i];
}

// ---------------------------------------------------------------- projection
__global__ void proj_kernel(const float* __restrict__ x,
                            const float* __restrict__ latw,
                            const float* __restrict__ latb,
                            const float* __restrict__ condw,
                            const float* __restrict__ condb,
                            __bf16* __restrict__ out) {
  int idx = blockIdx.x * blockDim.x + threadIdx.x;   // over B*H
  int n = idx % H_SZ;
  int b = idx / H_SZ;
  const float* xr = x + (size_t)b * (D_INP + D_COND);
  float acc;
  if (n < H_SZ / 2) {
    acc = latb[n];
    const float* w = latw + (size_t)n * D_INP;
#pragma unroll 8
    for (int i = 0; i < D_INP; ++i) acc += xr[i] * w[i];
  } else {
    int n2 = n - H_SZ / 2;
    acc = condb[n2];
    const float* w = condw + (size_t)n2 * D_COND;
    for (int i = 0; i < D_COND; ++i) acc += xr[D_INP + i] * w[i];
  }
  out[(size_t)b * H_SZ + n] = (__bf16)acc;
}

// ---------------------------------------------------------------- GRU layer
#define GRU_BM 32
#define LDSP   8
#define XLD    (H_SZ + LDSP)   // padded LDS row stride (bf16 elems)

__global__ void __launch_bounds__(256)
gru_layer_kernel(const __bf16* __restrict__ xl_in,    // (B,H) bf16
                 const float*  __restrict__ h_prev,   // (B,H) f32
                 const __bf16* __restrict__ wih,      // (3H,H) bf16
                 const __bf16* __restrict__ whh,      // (3H,H) bf16
                 const float*  __restrict__ bih,      // (3H)
                 const float*  __restrict__ bhh,      // (3H)
                 float*  __restrict__ h_out,          // (B,H) f32 -> d_out
                 __bf16* __restrict__ xl_out,         // next-layer input
                 int out_ld) {
  __shared__ __align__(16) __bf16 sX[GRU_BM * XLD];
  __shared__ __align__(16) __bf16 sH[GRU_BM * XLD];
  const int tid = threadIdx.x;
  const int b0  = blockIdx.x * GRU_BM;

  // stage xl (bf16 copy) and h_prev (f32 -> bf16) into LDS
  {
    const __bf16* gx = xl_in + (size_t)b0 * H_SZ;
    const float*  gh = h_prev + (size_t)b0 * H_SZ;
    for (int i = tid * 4; i < GRU_BM * H_SZ; i += 256 * 4) {
      int r = i / H_SZ, c = i % H_SZ;
      *(uint2*)&sX[r * XLD + c] = *(const uint2*)&gx[i];
      float4 f = *(const float4*)&gh[i];
      union { __bf16 h4[4]; uint2 u; } cv;
      cv.h4[0] = (__bf16)f.x; cv.h4[1] = (__bf16)f.y;
      cv.h4[2] = (__bf16)f.z; cv.h4[3] = (__bf16)f.w;
      *(uint2*)&sH[r * XLD + c] = cv.u;
    }
  }
  __syncthreads();

  const int wv = tid >> 5;
  const int mt = wv & 1;        // 2 M-tiles of 16 rows
  const int ng = wv >> 1;       // 4 N-groups
  const int lane = tid & 31;
  const int hf = lane >> 4;
  const int ln = lane & 15;

  for (int nt = ng; nt < H_SZ / 16; nt += 4) {
    const int n0 = nt * 16;
    v8f accR = {}, accZ = {}, accI = {}, accN = {};
    for (int k0 = 0; k0 < H_SZ; k0 += 32) {
      v16bf ax = loadA16(sX, XLD, mt * 16, k0);
      v16bf ah = loadA16(sH, XLD, mt * 16, k0);
      v16bf br0 = loadB16(wih, H_SZ, 0 * H_SZ + n0, k0);
      accR = wmma_bf16(ax, br0, accR);
      v16bf br1 = loadB16(whh, H_SZ, 0 * H_SZ + n0, k0);
      accR = wmma_bf16(ah, br1, accR);
      v16bf bz0 = loadB16(wih, H_SZ, 1 * H_SZ + n0, k0);
      accZ = wmma_bf16(ax, bz0, accZ);
      v16bf bz1 = loadB16(whh, H_SZ, 1 * H_SZ + n0, k0);
      accZ = wmma_bf16(ah, bz1, accZ);
      v16bf bn0 = loadB16(wih, H_SZ, 2 * H_SZ + n0, k0);
      accI = wmma_bf16(ax, bn0, accI);
      v16bf bn1 = loadB16(whh, H_SZ, 2 * H_SZ + n0, k0);
      accN = wmma_bf16(ah, bn1, accN);
    }
    const int nc = n0 + ln;
    const float br  = bih[nc] + bhh[nc];
    const float bz  = bih[H_SZ + nc] + bhh[H_SZ + nc];
    const float bin = bih[2 * H_SZ + nc];
    const float bhn = bhh[2 * H_SZ + nc];
#pragma unroll
    for (int v = 0; v < 8; ++v) {
      const int m = mt * 16 + v + 8 * hf;
      const size_t row = (size_t)(b0 + m);
      float r  = sigmoidf_(accR[v] + br);
      float z  = sigmoidf_(accZ[v] + bz);
      float nn = tanhf(accI[v] + bin + r * (accN[v] + bhn));
      float hp = h_prev[row * H_SZ + nc];
      float hn = (1.0f - z) * nn + z * hp;
      h_out[row * H_SZ + nc] = hn;
      xl_out[row * out_ld + nc] = (__bf16)hn;
    }
  }
}

// ---------------------------------------------------------------- decoder step
#define DEC_BM 16
#define DLD (DEC_IN + 8)    // 1160 padded row stride for dec_in tile
#define WLD (K_SZ + 8)      // 1032 padded row stride for softmax weights

__global__ void __launch_bounds__(256)
dec_step_kernel(const __bf16* __restrict__ din_g,   // (B,1152) bf16
                __bf16* __restrict__ din_gw,        // same buffer, cum cols
                const __bf16* __restrict__ dwq,     // (1024,1152) bf16
                const float*  __restrict__ dbq,     // (1024)
                const __bf16* __restrict__ etq,     // (128,1024) bf16 (E^T)
                float* __restrict__ cum,            // (B,128) ws
                float* __restrict__ stepbuf,        // (B,128) ws
                float* __restrict__ logits_out,     // (B,1024) slice of d_out
                float* __restrict__ step_out,       // (B,128)  slice of d_out
                int q) {
  __shared__ __align__(16) char smem[DEC_BM * K_SZ * 4 + DEC_BM * DLD * 2];
  float*  sL = (float*)smem;                              // 16x1024 f32 logits
  __bf16* sD = (__bf16*)(smem + DEC_BM * K_SZ * 4);       // dec_in tile
  __bf16* sW = (__bf16*)(smem + DEC_BM * K_SZ * 4);       // softmax w (aliases sD)
  const int tid = threadIdx.x;
  const int b0  = blockIdx.x * DEC_BM;

  // phase 1: stage dec_in tile (zero cum columns at q==0)
  {
    const __bf16* g = din_g + (size_t)b0 * DEC_IN;
    for (int i = tid * 4; i < DEC_BM * DEC_IN; i += 256 * 4) {
      int r = i / DEC_IN, c = i % DEC_IN;
      uint2 d;
      if (q == 0 && c >= H_SZ) d = make_uint2(0u, 0u);
      else d = *(const uint2*)&g[i];
      *(uint2*)&sD[r * DLD + c] = d;
    }
  }
  __syncthreads();

  const int wv = tid >> 5, lane = tid & 31, hf = lane >> 4, ln = lane & 15;

  // phase 2: logits = dec_in @ dec_w^T + b  (16 x 1024, K=1152)
  for (int nt = wv; nt < K_SZ / 16; nt += 8) {
    const int n0 = nt * 16;
    v8f acc = {};
    for (int k0 = 0; k0 < DEC_IN; k0 += 32) {
      v16bf a = loadA16(sD, DLD, 0, k0);
      v16bf b = loadB16(dwq, DEC_IN, n0, k0);
      acc = wmma_bf16(a, b, acc);
    }
    const int nc = n0 + ln;
    const float bias = dbq[nc];
#pragma unroll
    for (int v = 0; v < 8; ++v) {
      const int m = v + 8 * hf;
      float lg = acc[v] + bias;
      sL[m * K_SZ + nc] = lg;
      logits_out[(size_t)(b0 + m) * K_SZ + nc] = lg;
    }
  }
  __syncthreads();

  // phase 3: row softmax over K=1024, 2 rows per wave, wave32 shfl reductions
  for (int rr = 0; rr < 2; ++rr) {
    const int r = wv * 2 + rr;
    const float* row = sL + r * K_SZ;
    float mx = -3.4e38f;
    for (int c = lane; c < K_SZ; c += 32) mx = fmaxf(mx, row[c]);
    for (int off = 16; off; off >>= 1) mx = fmaxf(mx, __shfl_xor(mx, off));
    float sum = 0.f;
    for (int c = lane; c < K_SZ; c += 32) sum += __expf((row[c] - mx) * INV_TAU);
    for (int off = 16; off; off >>= 1) sum += __shfl_xor(sum, off);
    const float inv = 1.0f / sum;
    for (int c = lane; c < K_SZ; c += 32)
      sW[r * WLD + c] = (__bf16)(__expf((row[c] - mx) * INV_TAU) * inv);
  }
  __syncthreads();

  // phase 4: e_q = w @ E_eff  (16 x 128, K=1024) + cum/step update
  {
    const int n0 = wv * 16;   // 8 waves cover 128 cols
    v8f acc = {};
    for (int k0 = 0; k0 < K_SZ; k0 += 32) {
      v16bf a = loadA16(sW, WLD, 0, k0);
      v16bf b = loadB16(etq, K_SZ, n0, k0);
      acc = wmma_bf16(a, b, acc);
    }
    const int nc = n0 + ln;
#pragma unroll
    for (int v = 0; v < 8; ++v) {
      const int m = v + 8 * hf;
      const size_t row = (size_t)(b0 + m);
      float e  = acc[v];
      float c0 = (q == 0) ? 0.f : cum[row * D_INP + nc];
      float s0 = (q == 0) ? 0.f : stepbuf[row * D_INP + nc];
      float c1 = c0 + e;
      float s1 = s0 + e;
      stepbuf[row * D_INP + nc] = s1;
      if (q < NQ - 1) {
        cum[row * D_INP + nc] = c1;
        din_gw[row * DEC_IN + H_SZ + nc] = (__bf16)c1;
      } else {
        step_out[row * D_INP + nc] = s1;
      }
    }
  }
}

// ---------------------------------------------------------------- launcher
extern "C" void kernel_launch(void* const* d_in, const int* in_sizes, int n_in,
                              void* d_out, int out_size, void* d_ws, size_t ws_size,
                              hipStream_t stream) {
  (void)in_sizes; (void)n_in; (void)out_size; (void)ws_size;
  const float* x      = (const float*)d_in[0];
  const float* hidden = (const float*)d_in[1];
  const float* lat_w  = (const float*)d_in[2];
  const float* lat_b  = (const float*)d_in[3];
  const float* cond_w = (const float*)d_in[4];
  const float* cond_b = (const float*)d_in[5];
  const float* gwih   = (const float*)d_in[6];
  const float* gwhh   = (const float*)d_in[7];
  const float* gbih   = (const float*)d_in[8];
  const float* gbhh   = (const float*)d_in[9];
  const float* dec_w  = (const float*)d_in[10];
  const float* dec_b  = (const float*)d_in[11];
  const float* E      = (const float*)d_in[12];

  float* out        = (float*)d_out;
  float* logits     = out;                                      // (NQ,B,K)
  float* hidden_out = out + (size_t)NQ * B_SZ * K_SZ;           // (L,B,H)
  float* step_out   = hidden_out + (size_t)L_SZ * B_SZ * H_SZ;  // (B,128)

  // workspace layout (bf16 region first, f32 region after; ~126 MB total)
  const size_t GRU_W = (size_t)L_SZ * 3 * H_SZ * H_SZ;
  __bf16* gih = (__bf16*)d_ws;
  __bf16* ghh = gih + GRU_W;
  __bf16* dwb = ghh + GRU_W;
  __bf16* etb = dwb + (size_t)NQ * K_SZ * DEC_IN;
  __bf16* xlA = etb + (size_t)NQ * D_INP * K_SZ;
  __bf16* xlB = xlA + (size_t)B_SZ * H_SZ;
  __bf16* dnb = xlB + (size_t)B_SZ * H_SZ;
  float*  cum = (float*)(dnb + (size_t)B_SZ * DEC_IN);
  float*  stp = cum + (size_t)B_SZ * D_INP;

  // weight precision conversion (one pass; L2-resident afterwards)
  cvt_bf16_kernel<<<2048, 256, 0, stream>>>(gwih, gih, GRU_W);
  cvt_bf16_kernel<<<2048, 256, 0, stream>>>(gwhh, ghh, GRU_W);
  cvt_bf16_kernel<<<2048, 256, 0, stream>>>(dec_w, dwb, (size_t)NQ * K_SZ * DEC_IN);
  transE_kernel<<<(NQ * K_SZ * D_INP + 255) / 256, 256, 0, stream>>>(E, etb);

  // input projection -> xlA
  proj_kernel<<<(B_SZ * H_SZ) / 256, 256, 0, stream>>>(x, lat_w, lat_b, cond_w,
                                                       cond_b, xlA);

  // 4 sequential GRU layers; last layer writes h_out into dec_in buffer (ld=1152)
  const __bf16* xin = xlA;
  for (int l = 0; l < L_SZ; ++l) {
    __bf16* xout;
    int old_ld;
    if (l == L_SZ - 1) { xout = dnb; old_ld = DEC_IN; }
    else               { xout = (l & 1) ? xlA : xlB; old_ld = H_SZ; }
    gru_layer_kernel<<<B_SZ / GRU_BM, 256, 0, stream>>>(
        xin, hidden + (size_t)l * B_SZ * H_SZ,
        gih + (size_t)l * 3 * H_SZ * H_SZ, ghh + (size_t)l * 3 * H_SZ * H_SZ,
        gbih + (size_t)l * 3 * H_SZ, gbhh + (size_t)l * 3 * H_SZ,
        hidden_out + (size_t)l * B_SZ * H_SZ, xout, old_ld);
    xin = xout;
  }

  // 8 sequential soft-RVQ steps (logits GEMM + softmax + codebook GEMM fused)
  for (int q = 0; q < NQ; ++q) {
    dec_step_kernel<<<B_SZ / DEC_BM, 256, 0, stream>>>(
        dnb, dnb, dwb + (size_t)q * K_SZ * DEC_IN, dec_b + (size_t)q * K_SZ,
        etb + (size_t)q * D_INP * K_SZ, cum, stp,
        logits + (size_t)q * B_SZ * K_SZ, step_out, q);
  }
}
